// OptimizedLSTM_26491358281815
// MI455X (gfx1250) — matrix operand order repaired and checked by reading, not compiled, not run on hardware
//
#include <hip/hip_runtime.h>

// LSTM-scan fused persistent kernel for gfx1250 (MI455X).
// Key insight: the scan recurrence is elementwise in the batch (S) axis,
// so each 16-row S-block evolves independently. One workgroup (16 waves,
// 512 threads) owns one S-block and ALL 16 H-column tiles: h-state lives
// entirely in LDS (double-buffered), the per-step sync is a single
// workgroup barrier, and there is NO device-wide synchronization at all.
// x rows are staged per-WGP with GLOBAL_LOAD_ASYNC_TO_LDS_B128 (ASYNCcnt),
// double-buffered to overlap the next step's copy with this step's WMMAs.
// c-state stays in the WMMA accumulator registers for the whole scan.

typedef __bf16 bf16;
typedef __attribute__((ext_vector_type(16))) __bf16 v16bf;
typedef __attribute__((ext_vector_type(8)))  __bf16 v8bf;
typedef __attribute__((ext_vector_type(8)))  float  v8f;
typedef __attribute__((ext_vector_type(4)))  float  v4f;

#define T_STEPS 1024   // reference "B" (scan axis)
#define S_DIM   256    // per-step batch
#define F_DIM   256
#define H_DIM   256
#define NBLOCKS 16     // one workgroup per independent S-block
#define NTHREADS 512   // 16 waves: wave w owns H-column block w

// Async global->LDS copy of one 16B chunk per lane (GV mode, ASYNCcnt).
// LDS operand is the low 32 bits of the flat shared-aperture address,
// which per the aperture rules IS the LDS byte offset.
#define ASYNC_LDS_COPY16(gptr, lptr)                                        \
  asm volatile("global_load_async_to_lds_b128 %0, %1, off" ::               \
                   "v"((unsigned int)(uintptr_t)(lptr)),                    \
                   "v"(gptr)                                                \
               : "memory")
#define WAIT_ASYNC0() asm volatile("s_wait_asynccnt 0x0" ::: "memory")

// ---------------- prep: transpose+convert weights to bf16 ------------------
__global__ void lstm_prep(const float* __restrict__ wih,
                          const float* __restrict__ whh,
                          bf16* __restrict__ wih_t,
                          bf16* __restrict__ whh_t) {
  const int idx0   = blockIdx.x * blockDim.x + threadIdx.x;
  const int stride = gridDim.x * blockDim.x;
  const int W = 4 * H_DIM * F_DIM;                        // 1024*256
  for (int idx = idx0; idx < W; idx += stride) {
    int n = idx >> 8;       // gate-column (row of transposed weight)
    int k = idx & 255;      // K index (contiguous in transposed layout)
    wih_t[idx] = (bf16)wih[k * (4 * H_DIM) + n];
    whh_t[idx] = (bf16)whh[k * (4 * H_DIM) + n];
  }
}

// ---------------- helpers --------------------------------------------------
__device__ __forceinline__ float fsigmoid(float x) {
  return 1.0f / (1.0f + __expf(-x));
}
__device__ __forceinline__ float ftanh(float x) {
  return 2.0f / (1.0f + __expf(-2.0f * x)) - 1.0f;
}

// ---------------- main scan kernel -----------------------------------------
__global__ void __launch_bounds__(NTHREADS, 1)
lstm_scan(const float* __restrict__ x,        // (T, S, F) fp32
          const float* __restrict__ bias,     // (4H) fp32
          const bf16*  __restrict__ wih_t,    // (4H, F) bf16, K-contiguous
          const bf16*  __restrict__ whh_t,    // (4H, H) bf16, K-contiguous
          float* __restrict__ out) {          // hidden_seq | h_T | c_T
  const int tid  = threadIdx.x;
  const int lane = tid & 31;
  const int n    = tid >> 5;                  // H-block (= wave id, 0..15)
  const int mb   = blockIdx.x;                // S-block (0..15)
  const int half = lane >> 4;                 // K-half selector
  const int col  = lane & 15;                 // N column / A row (M)

  // LDS: x rows double-buffered (fp32, async-staged), h state double-
  // buffered (bf16, written directly by the producing waves).
  __shared__ __align__(16) float x_stage[2][16 * F_DIM];   // 2 x 16 KB
  __shared__ __align__(16) bf16  h_stage[2][16 * H_DIM];   // 2 x  8 KB

  // h(0) = 0
  for (int i = tid; i < 16 * H_DIM; i += NTHREADS)
    h_stage[0][i] = (bf16)0.0f;

  // Per-gate bias (same column for all 8 accumulator rows of this lane).
  float bgate[4];
#pragma unroll
  for (int g = 0; g < 4; ++g) bgate[g] = bias[g * H_DIM + n * 16 + col];

  // Per-gate B-matrix row base (transposed weights: row = gate column).
  const bf16* wihB[4];
  const bf16* whhB[4];
#pragma unroll
  for (int g = 0; g < 4; ++g) {
    const int gcol = g * H_DIM + n * 16 + col;
    wihB[g] = wih_t + (size_t)gcol * F_DIM;
    whhB[g] = whh_t + (size_t)gcol * H_DIM;
  }

  // Cell state tile lives in registers for the entire scan.
  v8f c_tile;
#pragma unroll
  for (int j = 0; j < 8; ++j) c_tile[j] = 0.0f;

  // Prologue: async-stage x(t=0) rows (16 x 256 fp32 = 16 KB = 1024 chunks).
  {
    const float* gx = x + (size_t)mb * 16 * F_DIM;
#pragma unroll
    for (int kch = 0; kch < 2; ++kch) {
      const int idx = tid + kch * NTHREADS;            // 16-byte chunk id
      ASYNC_LDS_COPY16(gx + idx * 4, &x_stage[0][idx * 4]);
    }
  }

  for (int t = 0; t < T_STEPS; ++t) {
    // Own async x-copy done, then barrier: all waves' copies + all h
    // writes from step t-1 are visible.
    WAIT_ASYNC0();
    __syncthreads();

    // Accumulators start from bias.
    v8f acc[4];
#pragma unroll
    for (int g = 0; g < 4; ++g)
#pragma unroll
      for (int j = 0; j < 8; ++j) acc[g][j] = bgate[g];

    // ---- input projection: x_t(row) @ W_ih from LDS, K = F = 256 ----
    // (redundant per-wave cvt is VALU and co-executes with the XDL pipe)
    const float* __restrict__ xr = &x_stage[t & 1][col * F_DIM];
#pragma unroll
    for (int kb = 0; kb < F_DIM / 32; ++kb) {
      const int k0 = kb * 32 + half * 16;
      // A fragment: K {k0..k0+15} (contiguous, matches B fragment)
      v4f f0 = *(const v4f*)(xr + k0);
      v4f f1 = *(const v4f*)(xr + k0 + 4);
      v4f f2 = *(const v4f*)(xr + k0 + 8);
      v4f f3 = *(const v4f*)(xr + k0 + 12);
      v16bf afrag;
#pragma unroll
      for (int j = 0; j < 4; ++j) {
        afrag[j]      = (bf16)f0[j];
        afrag[4 + j]  = (bf16)f1[j];
        afrag[8 + j]  = (bf16)f2[j];
        afrag[12 + j] = (bf16)f3[j];
      }
#pragma unroll
      for (int g = 0; g < 4; ++g) {
        // B fragment: 16 contiguous K values for this lane's column;
        // weights are ~1 MB total and stay resident in the 192 MB L2.
        v16bf bfrag = *(const v16bf*)(wihB[g] + kb * 32 + half * 16);
        acc[g] = __builtin_amdgcn_wmma_f32_16x16x32_bf16(
            false, afrag, false, bfrag, (short)0, acc[g], false, false);
      }
    }

    // ---- recurrent projection: h(row) @ W_hh from LDS, K = H = 256 ----
    const bf16* __restrict__ hr = &h_stage[t & 1][col * H_DIM];
#pragma unroll
    for (int kb = 0; kb < H_DIM / 32; ++kb) {
      const int k0 = kb * 32 + half * 16;
      v8bf lo = *(const v8bf*)(hr + k0);
      v8bf hi = *(const v8bf*)(hr + k0 + 8);
      v16bf afrag;
#pragma unroll
      for (int j = 0; j < 8; ++j) {
        afrag[j]     = lo[j];
        afrag[8 + j] = hi[j];
      }
#pragma unroll
      for (int g = 0; g < 4; ++g) {
        v16bf bfrag = *(const v16bf*)(whhB[g] + kb * 32 + half * 16);
        acc[g] = __builtin_amdgcn_wmma_f32_16x16x32_bf16(
            false, afrag, false, bfrag, (short)0, acc[g], false, false);
      }
    }

    // Overlap: async-copy next step's x block into the other LDS buffer
    // (last read in the previous step; every wave is past that read after the
    // barrier above), and warm L2 two steps ahead.
    if (t + 1 < T_STEPS) {
      const float* gx = x + ((size_t)(t + 1) * S_DIM + mb * 16) * F_DIM;
#pragma unroll
      for (int kch = 0; kch < 2; ++kch) {
        const int idx = tid + kch * NTHREADS;
        ASYNC_LDS_COPY16(gx + idx * 4, &x_stage[(t + 1) & 1][idx * 4]);
      }
    }
    if (t + 2 < T_STEPS)
      __builtin_prefetch(
          x + ((size_t)(t + 2) * S_DIM + mb * 16) * F_DIM + tid * 8, 0, 0);

    // ---- elementwise LSTM cell update + writes ----
    // C/D layout: element j -> row (j + 8*half) of tile, column `col`.
    bf16* __restrict__ hnext = h_stage[(t + 1) & 1];
#pragma unroll
    for (int j = 0; j < 8; ++j) {
      float iv = fsigmoid(acc[0][j]);
      float fv = fsigmoid(acc[1][j]);
      float gv = ftanh(acc[2][j]);
      float ov = fsigmoid(acc[3][j]);
      float cn = fv * c_tile[j] + iv * gv;
      c_tile[j] = cn;
      float hn = ov * ftanh(cn);

      const int sl   = j + half * 8;            // row within S-block
      const int s    = mb * 16 + sl;
      const int hidx = n * 16 + col;
      out[(size_t)t * (S_DIM * H_DIM) + (size_t)s * H_DIM + hidx] = hn;
      hnext[sl * H_DIM + hidx] = (bf16)hn;      // ds_store: next-step state
      if (t == T_STEPS - 1) {
        const size_t tail = (size_t)T_STEPS * S_DIM * H_DIM;
        out[tail + (size_t)s * H_DIM + hidx] = hn;                         // h_T
        out[tail + (size_t)S_DIM * H_DIM + (size_t)s * H_DIM + hidx] = cn; // c_T
      }
    }
    // Visibility of hnext for all waves is provided by the barrier at the
    // top of the next iteration.
  }
}

// ---------------- host side ------------------------------------------------
extern "C" void kernel_launch(void* const* d_in, const int* in_sizes, int n_in,
                              void* d_out, int out_size, void* d_ws,
                              size_t ws_size, hipStream_t stream) {
  const float* x    = (const float*)d_in[0];   // (1024, 256, 256)
  const float* wih  = (const float*)d_in[1];   // (256, 1024)
  const float* whh  = (const float*)d_in[2];   // (256, 1024)
  const float* bias = (const float*)d_in[3];   // (1024,)
  float* out = (float*)d_out;

  char* ws = (char*)d_ws;
  const size_t wbytes = (size_t)4 * H_DIM * F_DIM * sizeof(bf16); // 512 KB
  bf16* wih_t = (bf16*)(ws);
  bf16* whh_t = (bf16*)(ws + wbytes);

  lstm_prep<<<256, 256, 0, stream>>>(wih, whh, wih_t, whh_t);
  lstm_scan<<<NBLOCKS, NTHREADS, 0, stream>>>(x, bias, wih_t, whh_t, out);
}
